// PyGRUCell_74758200754588
// MI455X (gfx1250) — matrix operand order repaired and checked by reading, not compiled, not run on hardware
//
#include <hip/hip_runtime.h>
#include <math.h>

// GRU: B=32, L=1024, D=H=512
#define Bq 32
#define Lq 1024
#define Dq 512
#define Hq 512
#define NGq 1536          // 3*H (z,r,h gate columns concatenated)
#define GRIDSZ 16         // persistent blocks in the scan kernel

typedef __attribute__((ext_vector_type(16))) _Float16 v16h;
typedef __attribute__((ext_vector_type(8)))  _Float16 h8;
typedef __attribute__((ext_vector_type(8)))  float    v8f;
typedef __attribute__((ext_vector_type(4)))  float    f4;
typedef __attribute__((ext_vector_type(4)))  unsigned int u32x4;
typedef __attribute__((ext_vector_type(8)))  int      i32x8;
typedef __attribute__((ext_vector_type(4)))  int      i32x4;

// ---------------- workspace layout (bytes) ----------------
static const size_t WXT_OFF   = 0;                                   // f16 [1536][512]  W_x^T (z,r,h)
static const size_t WHT_OFF   = WXT_OFF + (size_t)NGq * Dq * 2;      // f16 [1536][512]  W_h^T (z,r,h)
static const size_t GATES_OFF = WHT_OFF + (size_t)NGq * Hq * 2;      // f32 [3][L][B][H] pre-activations
static const size_t HBUF_OFF  = GATES_OFF + (size_t)3 * Lq * Bq * Hq * 4; // f32 [B][H]
static const size_t RH_OFF    = HBUF_OFF + (size_t)Bq * Hq * 4;      // f16 [B][H]  r*h
static const size_t BAR_OFF   = RH_OFF + (size_t)Bq * Hq * 2;        // u32 grid barrier

__device__ __forceinline__ v8f wmma_f16(v16h a, v16h b, v8f c) {
  // (neg_a, A, neg_b, B, c_mod, C, reuse_a, reuse_b)
  return __builtin_amdgcn_wmma_f32_16x16x32_f16(false, a, false, b, (short)0, c, false, false);
}

// Build a 16x32 f16 A-fragment from an fp32 row (contiguous layout: lanes 0-15 hold
// K {0..15}, lanes 16-31 hold K {16..31} of the 32-wide K block).
__device__ __forceinline__ v16h a_frag_from_f32(const float* rowp, int kb, int hi) {
  const float* p = rowp + kb + hi * 16;
  f4 a0 = *(const f4*)(p);
  f4 a1 = *(const f4*)(p + 4);
  f4 a2 = *(const f4*)(p + 8);
  f4 a3 = *(const f4*)(p + 12);
  v16h a;
  a[0] = (_Float16)a0[0];  a[1] = (_Float16)a0[1];  a[2] = (_Float16)a0[2];  a[3] = (_Float16)a0[3];
  a[4] = (_Float16)a1[0];  a[5] = (_Float16)a1[1];  a[6] = (_Float16)a1[2];  a[7] = (_Float16)a1[3];
  a[8] = (_Float16)a2[0];  a[9] = (_Float16)a2[1];  a[10] = (_Float16)a2[2]; a[11] = (_Float16)a2[3];
  a[12] = (_Float16)a3[0]; a[13] = (_Float16)a3[1]; a[14] = (_Float16)a3[2]; a[15] = (_Float16)a3[3];
  return a;
}

__device__ __forceinline__ v16h a_frag_from_f16(const _Float16* rowp, int kb, int hi) {
  h8 lo = *(const h8*)(rowp + kb + hi * 16);
  h8 hh = *(const h8*)(rowp + kb + hi * 16 + 8);
  return __builtin_shufflevector(lo, hh, 0, 1, 2, 3, 4, 5, 6, 7, 8, 9, 10, 11, 12, 13, 14, 15);
}

// ---------------- TDM: 1-D contiguous tile, global -> LDS ----------------
// Builds the Tensor DMA Descriptor (D#) per CDNA5 ISA 8.3/8.4:
//  group0: count=1, lds_addr[63:32], global_addr[120:64], type=2
//  group1: data_size[17:16], tensor_dim0[79:48], tensor_dim1[111:80]=1,
//          tile_dim0[127:112], tensor_dim0_stride[207:160]
// groups 2/3 zero (tensor <= 2D).
__device__ __forceinline__ void tdm_load_1d(unsigned lds_addr, const void* gptr,
                                            unsigned nelem, unsigned data_size_code) {
  unsigned long long ga = (unsigned long long)gptr;
  u32x4 g0;
  g0[0] = 1u;                                               // count=1
  g0[1] = lds_addr;                                         // LDS byte address
  g0[2] = (unsigned)(ga & 0xFFFFFFFFu);                     // global_addr lo
  g0[3] = (unsigned)((ga >> 32) & 0x01FFFFFFu) | (2u << 30);// global_addr hi | type=2
  i32x8 g1;
  g1[0] = (int)(data_size_code << 16);                      // wg_mask=0, data_size
  g1[1] = (int)((nelem & 0xFFFFu) << 16);                   // tensor_dim0[15:0]
  g1[2] = (int)((nelem >> 16) & 0xFFFFu) | (1 << 16);       // tensor_dim0[31:16], tensor_dim1=1 (lo)
  g1[3] = (int)((nelem & 0xFFFFu) << 16);                   // tensor_dim1 hi=0, tile_dim0
  g1[4] = 0;                                                // tile_dim1=0 (unused), tile_dim2=0
  g1[5] = (int)nelem;                                       // tensor_dim0_stride lo32
  g1[6] = 0;                                                // stride hi, dim1_stride lo
  g1[7] = 0;
  i32x4 z4 = {0, 0, 0, 0};
#if __clang_major__ >= 23
  i32x8 z8 = {0, 0, 0, 0, 0, 0, 0, 0};
  __builtin_amdgcn_tensor_load_to_lds(g0, g1, z4, z4, z8, 0);
#else
  __builtin_amdgcn_tensor_load_to_lds(g0, g1, z4, z4, 0);
#endif
}

// ---------------- kernel 0: transpose + convert weights to f16 ----------------
__global__ __launch_bounds__(256) void cvt_w(
    const float* __restrict__ Wxz, const float* __restrict__ Wxr, const float* __restrict__ Wxh,
    const float* __restrict__ Whz, const float* __restrict__ Whr, const float* __restrict__ Whh,
    _Float16* __restrict__ wxt, _Float16* __restrict__ wht) {
  const int mat = blockIdx.y;
  const float* src;
  _Float16* dst;
  switch (mat) {
    case 0:  src = Wxz; dst = wxt;                        break;
    case 1:  src = Wxr; dst = wxt + (size_t)Hq * Dq;      break;
    case 2:  src = Wxh; dst = wxt + (size_t)2 * Hq * Dq;  break;
    case 3:  src = Whz; dst = wht;                        break;
    case 4:  src = Whr; dst = wht + (size_t)Hq * Hq;      break;
    default: src = Whh; dst = wht + (size_t)2 * Hq * Hq;  break;
  }
  const int idx = blockIdx.x * 256 + threadIdx.x;  // = k*512 + n (coalesced read)
  const int k = idx >> 9;
  const int n = idx & 511;
  dst[(size_t)n * Dq + k] = (_Float16)src[idx];    // dst[n][k] = W[k][n]
}

// ---------------- kernel 1: fused input projections ----------------
// C[m, ncol] = x[m,:] @ Wx[:, ncol] + bias,  m = b*L + l, ncol in [0,1536)
// stored as gates[g][l][b][h] fp32. Block tile 64(M) x 64(N), 4 waves,
// each wave does a 16x64 strip (1 A frag reused across 4 B frags per K step).
// Memory-bound (~256MB @ 23.3 TB/s); x is L2-resident (64MB << 192MB L2), so
// fragments come straight from global with no LDS staging.
__global__ __launch_bounds__(128) void gemm_x(
    const float* __restrict__ x, const _Float16* __restrict__ wxt,
    const float* __restrict__ bz, const float* __restrict__ br,
    const float* __restrict__ bh, float* __restrict__ gates) {
  const int tid = threadIdx.x;
  const int lane = tid & 31;
  const int wv = tid >> 5;
  const int r16 = lane & 15;
  const int hi = lane >> 4;
  const int nbase = blockIdx.x * 64;
  const int mbase = blockIdx.y * 64 + wv * 16;

  const float* xrow = x + (size_t)(mbase + r16) * Dq;

  v8f acc[4] = {v8f{}, v8f{}, v8f{}, v8f{}};
  for (int kb = 0; kb < Dq; kb += 32) {
    v16h a = a_frag_from_f32(xrow, kb, hi);
#pragma unroll
    for (int nt = 0; nt < 4; ++nt) {
      const _Float16* wp = wxt + (size_t)(nbase + nt * 16 + r16) * Dq + kb + hi * 16;
      v16h b = *(const v16h*)wp;
      acc[nt] = wmma_f16(a, b, acc[nt]);
    }
  }

  const int g = nbase >> 9;  // 64-wide tiles never straddle a gate boundary
  const float* bias = (g == 0) ? bz : ((g == 1) ? br : bh);
#pragma unroll
  for (int nt = 0; nt < 4; ++nt) {
    const int col = nbase + nt * 16 + r16;
    const int hcol = col & (Hq - 1);
    const float bv = bias[hcol];
#pragma unroll
    for (int j = 0; j < 8; ++j) {
      const int m = mbase + 8 * hi + j;     // C layout: row = 8*(lane>=16)+j
      const int b_ = m >> 10;               // m / L
      const int l_ = m & (Lq - 1);
      gates[(((size_t)g * Lq + l_) * Bq + b_) * Hq + hcol] = acc[nt][j] + bv;
    }
  }
}

// ---------------- grid barrier (persistent blocks) ----------------
__device__ __forceinline__ void grid_barrier(unsigned* bar, unsigned target) {
  __threadfence();
  __syncthreads();
  if (threadIdx.x == 0) {
    __hip_atomic_fetch_add(bar, 1u, __ATOMIC_ACQ_REL, __HIP_MEMORY_SCOPE_AGENT);
    while (__hip_atomic_load(bar, __ATOMIC_ACQUIRE, __HIP_MEMORY_SCOPE_AGENT) < target) {
      __builtin_amdgcn_s_sleep(1);
    }
  }
  __syncthreads();
  __threadfence();
}

// ---------------- kernel 2: sequential GRU scan ----------------
// 16 persistent blocks; block k owns output columns [32k, 32k+32).
// 4 waves per block: wave = (mi, ni) covering 16x16 tile of the 32x32 result.
// W_h slices (96KB f16, contiguous per gate in the W^T layout) are DMA'd into
// LDS once by the Tensor Data Mover; every B-fragment for all 1024 steps then
// comes from LDS instead of L2 (removes ~1.5GB of repeated L2 traffic and L2
// latency from the serial critical path).
__global__ __launch_bounds__(128) void gru_scan(
    const float* __restrict__ h0, const _Float16* __restrict__ wht,
    const float* __restrict__ gates, float* __restrict__ hbuf,
    _Float16* __restrict__ rhbuf, unsigned* __restrict__ bar,
    float* __restrict__ out, float* __restrict__ hT) {
  extern __shared__ __align__(128) char smem_raw[];
  _Float16* lds_w = (_Float16*)smem_raw;        // [3][32][512] f16 = 96KB

  const int tid = threadIdx.x;
  const int lane = tid & 31;
  const int wv = tid >> 5;
  const int mi = wv & 1;
  const int ni = wv >> 1;
  const int cbase = blockIdx.x * 32;
  const int r16 = lane & 15;
  const int hi = lane >> 4;
  const int col = cbase + 16 * ni + r16;   // this lane's output column

  // Kick off the TDM weight staging (one wave issues; overlaps h init).
  if (wv == 0) {
#pragma unroll
    for (int g = 0; g < 3; ++g) {
      const _Float16* src = wht + (size_t)(g * Hq + cbase) * Hq;   // 32 contig rows
      unsigned lds_off = (unsigned)(size_t)(void*)(lds_w + (size_t)g * 32 * Hq);
      tdm_load_1d(lds_off, src, 32 * Hq, /*data_size=2B*/ 1u);
    }
  }

  // init h from h_0 (each block its own column slice)
  for (int i = tid; i < 32 * 32; i += 128) {
    const int rr = i >> 5;
    const int cc = cbase + (i & 31);
    hbuf[rr * Hq + cc] = h0[rr * Hq + cc];
  }

  if (wv == 0) __builtin_amdgcn_s_wait_tensorcnt(0);
  __syncthreads();                       // LDS weights visible to all waves
  unsigned target = GRIDSZ;
  grid_barrier(bar, target);

  // B-fragment base pointers in LDS: row = gate-local column, contiguous K.
  const _Float16* wz = lds_w + (size_t)(0 * 32 + 16 * ni + r16) * Hq;
  const _Float16* wr = lds_w + (size_t)(1 * 32 + 16 * ni + r16) * Hq;
  const _Float16* wc = lds_w + (size_t)(2 * 32 + 16 * ni + r16) * Hq;

  const int arow = 16 * mi + r16;               // row for A-frag loads
  const float* ha = hbuf + (size_t)arow * Hq;
  const _Float16* rha = rhbuf + (size_t)arow * Hq;
  const int crow0 = 16 * mi + 8 * hi;           // C-layout base row for this lane

  float zreg[8], hreg[8];

  for (int t = 0; t < Lq; ++t) {
    // ---- pass 1: z, r for own columns; publish r*h ----
    v8f az = v8f{}, ar = v8f{};
    for (int kb = 0; kb < Hq; kb += 32) {
      v16h a = a_frag_from_f32(ha, kb, hi);                 // h (shared A for z and r)
      v16h bzf = *(const v16h*)(wz + kb + hi * 16);
      v16h brf = *(const v16h*)(wr + kb + hi * 16);
      az = wmma_f16(a, bzf, az);
      ar = wmma_f16(a, brf, ar);
    }
    const float* gz = gates + ((size_t)0 * Lq + t) * (Bq * Hq);
    const float* gr = gates + ((size_t)1 * Lq + t) * (Bq * Hq);
    const float* gh = gates + ((size_t)2 * Lq + t) * (Bq * Hq);
    __builtin_prefetch(gh + crow0 * Hq + col, 0, 3);        // pass-2 gate preacts
#pragma unroll
    for (int j = 0; j < 8; ++j) {
      const int row = crow0 + j;
      const float zp = az[j] + gz[row * Hq + col];
      const float rp = ar[j] + gr[row * Hq + col];
      const float z = 1.0f / (1.0f + __expf(-zp));
      const float r = 1.0f / (1.0f + __expf(-rp));
      const float hold = hbuf[row * Hq + col];
      zreg[j] = z;
      hreg[j] = hold;
      rhbuf[row * Hq + col] = (_Float16)(r * hold);
    }
    target += GRIDSZ;
    grid_barrier(bar, target);   // r*h fully published

    // ---- pass 2: candidate + state update ----
    v8f ac = v8f{};
    for (int kb = 0; kb < Hq; kb += 32) {
      v16h a = a_frag_from_f16(rha, kb, hi);
      v16h bcf = *(const v16h*)(wc + kb + hi * 16);
      ac = wmma_f16(a, bcf, ac);
    }
    if (t + 1 < Lq) {            // warm L2/L0 for next step's pass-1 gates
      const float* gz1 = gates + ((size_t)0 * Lq + t + 1) * (Bq * Hq);
      const float* gr1 = gates + ((size_t)1 * Lq + t + 1) * (Bq * Hq);
      __builtin_prefetch(gz1 + crow0 * Hq + col, 0, 3);
      __builtin_prefetch(gr1 + crow0 * Hq + col, 0, 3);
    }
#pragma unroll
    for (int j = 0; j < 8; ++j) {
      const int row = crow0 + j;
      const float hc = tanhf(ac[j] + gh[row * Hq + col]);
      const float hn = zreg[j] * hreg[j] + (1.0f - zreg[j]) * hc;
      hbuf[row * Hq + col] = hn;
      out[((size_t)row * Lq + t) * Hq + col] = hn;   // outputs[b][t][h]
      if (t == Lq - 1) hT[row * Hq + col] = hn;
    }
    target += GRIDSZ;
    grid_barrier(bar, target);   // h fully updated before next step
  }
}

extern "C" void kernel_launch(void* const* d_in, const int* in_sizes, int n_in,
                              void* d_out, int out_size, void* d_ws, size_t ws_size,
                              hipStream_t stream) {
  (void)in_sizes; (void)n_in; (void)out_size; (void)ws_size;
  const float* x   = (const float*)d_in[0];
  const float* h0  = (const float*)d_in[1];
  const float* Wxz = (const float*)d_in[2];
  const float* Whz = (const float*)d_in[3];
  const float* bz  = (const float*)d_in[4];
  const float* Wxr = (const float*)d_in[5];
  const float* Whr = (const float*)d_in[6];
  const float* br  = (const float*)d_in[7];
  const float* Wxh = (const float*)d_in[8];
  const float* Whh = (const float*)d_in[9];
  const float* bh  = (const float*)d_in[10];

  char* ws = (char*)d_ws;
  _Float16* wxt   = (_Float16*)(ws + WXT_OFF);
  _Float16* wht   = (_Float16*)(ws + WHT_OFF);
  float*    gates = (float*)(ws + GATES_OFF);
  float*    hbuf  = (float*)(ws + HBUF_OFF);
  _Float16* rhbuf = (_Float16*)(ws + RH_OFF);
  unsigned* bar   = (unsigned*)(ws + BAR_OFF);

  float* out = (float*)d_out;
  float* hT  = out + (size_t)Bq * Lq * Hq;

  hipMemsetAsync(bar, 0, 64, stream);  // graph-capture-safe barrier reset
  cvt_w<<<dim3(1024, 6), 256, 0, stream>>>(Wxz, Wxr, Wxh, Whz, Whr, Whh, wxt, wht);
  gemm_x<<<dim3(NGq / 64, (Bq * Lq) / 64), 128, 0, stream>>>(x, wxt, bz, br, bh, gates);
  gru_scan<<<GRIDSZ, 128, 3 * 32 * Hq * sizeof(_Float16), stream>>>(
      h0, wht, gates, hbuf, rhbuf, bar, out, hT);
}
